// BinaryConv2d_45603962748976
// MI455X (gfx1250) — matrix operand order repaired and checked by reading, not compiled, not run on hardware
//
#include <hip/hip_runtime.h>
#include <hip/hip_bf16.h>

typedef _Float16 h16;
typedef __attribute__((ext_vector_type(16))) _Float16 v16h;
typedef __attribute__((ext_vector_type(8)))  _Float16 v8h;
typedef __attribute__((ext_vector_type(8)))  float    v8f;

#define N_IMG 32
#define C_IN  128
#define H_IN  56
#define W_IN  56
#define K_OUT 256
#define H_PAD 58
#define KTOT  1152            // 128 * 9
#define HW    (H_IN * W_IN)   // 3136
#define PIX   (N_IMG * HW)    // 100352

// ---- pass 1: binarize + pack weights: w[256][128][3][3] f32 -> wpk[256][9][128] f16 ----
__global__ void pack_weights(const float* __restrict__ w, h16* __restrict__ wpk) {
    int idx = blockIdx.x * blockDim.x + threadIdx.x;   // 0 .. 294911
    int ch = idx / KTOT;
    int t  = idx % KTOT;
    int rs = t / C_IN;
    int c  = t % C_IN;
    int r  = rs / 3, s = rs % 3;
    float v  = w[((ch * C_IN + c) * 3 + r) * 3 + s];
    float sv = (v > 0.f) ? 1.f : ((v < 0.f) ? -1.f : 0.f);  // matches sign(0)=0
    wpk[idx] = (h16)sv;
}

// ---- pass 2: convert + zero-pad: x[32][128][56][56] f32 -> xp[32][58][58][128] f16 NHWC ----
__global__ void pad_convert(const float* __restrict__ x, h16* __restrict__ xp) {
    int idx = blockIdx.x * blockDim.x + threadIdx.x;   // 0 .. 13778943
    int c  = idx % C_IN;
    int t  = idx / C_IN;
    int iw = t % H_PAD; t /= H_PAD;
    int ih = t % H_PAD;
    int n  = t / H_PAD;
    int h = ih - 1, w = iw - 1;
    float v = 0.f;
    if (h >= 0 && h < H_IN && w >= 0 && w < W_IN)
        v = x[((n * C_IN + c) * H_IN + h) * W_IN + w];
    xp[idx] = (h16)v;
}

// ---- pass 3: implicit-GEMM conv via v_wmma_f32_16x16x32_f16 ----
// A = packed weights (M=channels), B = padded NHWC input (N=pixels), K = 1152.
// Wave tile: 32 channels x 64 pixels (2 A-frags x 4 B-subtiles = 8 accumulators).
// Per K-chunk: batch 4 A-loads + 4 B-loads up front, then 8 WMMAs, so the
// scheduler can clause the loads and use graduated s_wait_loadcnt.
__global__ __launch_bounds__(256)
void bconv_gemm(const h16* __restrict__ xp, const h16* __restrict__ wpk,
                const float* __restrict__ bias, float* __restrict__ out) {
    const int lane = threadIdx.x & 31;
    const int wave = threadIdx.x >> 5;
    const int l15  = lane & 15;
    const int hsel = lane >> 4;                 // 0: lanes 0-15, 1: lanes 16-31

    const int chBase = wave * 32;               // 8 waves * 32 ch = 256
    const int pxBase = blockIdx.x * 64;
    const int nimg   = pxBase / HW;             // block-uniform: 3136 % 64 == 0

    // A fragment bases (contiguous k = hsel*16 + s, same order as B): row M = l15, K offset hsel*16
    const h16* pa0 = wpk + (chBase      + l15) * KTOT + hsel * 16;
    const h16* pa1 = wpk + (chBase + 16 + l15) * KTOT + hsel * 16;

    // B fragment bases (contiguous k = hsel*16 + s, same order as A): N = l15, K offset hsel*16
    const h16* pb[4];
    int qs[4];
    #pragma unroll
    for (int j = 0; j < 4; ++j) {
        int p  = pxBase + j * 16 + l15;
        int q  = p - nimg * HW;
        int oh = q / W_IN;
        int ow = q - oh * W_IN;
        pb[j] = xp + ((nimg * H_PAD + oh) * H_PAD + ow) * C_IN + hsel * 16;
        qs[j] = q;
    }

    v8f acc0[4] = {};
    v8f acc1[4] = {};
    #pragma unroll
    for (int r = 0; r < 3; ++r) {
      #pragma unroll
      for (int s = 0; s < 3; ++s) {
        const int offA = (r * 3 + s) * C_IN;        // packed-weight K offset
        const int offB = (r * H_PAD + s) * C_IN;    // padded-NHWC K offset
        #pragma unroll
        for (int c0 = 0; c0 < C_IN; c0 += 32) {
          // ---- batched loads for this K-chunk ----
          v16h bf[4];
          #pragma unroll
          for (int j = 0; j < 4; ++j)
            bf[j] = *(const v16h*)(pb[j] + offB + c0);    // 32B contiguous each

          union { v16h v; v8h h[2]; } af0, af1;
          af0.h[0] = *(const v8h*)(pa0 + offA + c0);
          af0.h[1] = *(const v8h*)(pa0 + offA + c0 + 8);
          af1.h[0] = *(const v8h*)(pa1 + offA + c0);
          af1.h[1] = *(const v8h*)(pa1 + offA + c0 + 8);

          // ---- 8 WMMAs consuming the batch ----
          #pragma unroll
          for (int j = 0; j < 4; ++j) {
            acc0[j] = __builtin_amdgcn_wmma_f32_16x16x32_f16(
                false, af0.v, false, bf[j], (short)0, acc0[j], false, false);
            acc1[j] = __builtin_amdgcn_wmma_f32_16x16x32_f16(
                false, af1.v, false, bf[j], (short)0, acc1[j], false, false);
          }
        }
      }
    }

    // Hoisted bias (two 16-ch groups x 8 rows per lane)
    float bv0[8], bv1[8];
    #pragma unroll
    for (int i = 0; i < 8; ++i) {
        bv0[i] = bias[chBase      + i + hsel * 8];
        bv1[i] = bias[chBase + 16 + i + hsel * 8];
    }

    // Epilogue: D VGPR i -> M = i + hsel*8, N = l15 (consecutive pixels -> coalesced stores)
    #pragma unroll
    for (int j = 0; j < 4; ++j) {
      #pragma unroll
      for (int i = 0; i < 8; ++i) {
        int ch0 = chBase      + i + hsel * 8;
        int ch1 = chBase + 16 + i + hsel * 8;
        out[(nimg * K_OUT + ch0) * HW + qs[j]] = acc0[j][i] + bv0[i];
        out[(nimg * K_OUT + ch1) * HW + qs[j]] = acc1[j][i] + bv1[i];
      }
    }
}

extern "C" void kernel_launch(void* const* d_in, const int* in_sizes, int n_in,
                              void* d_out, int out_size, void* d_ws, size_t ws_size,
                              hipStream_t stream) {
    const float* x = (const float*)d_in[0];
    const float* w = (const float*)d_in[1];
    const float* b = (const float*)d_in[2];
    float* out = (float*)d_out;

    h16* wpk = (h16*)d_ws;                               // 576 KB
    h16* xp  = (h16*)((char*)d_ws + (1u << 20));         // 27.6 MB, 1 MB aligned

    pack_weights<<<(K_OUT * KTOT) / 256, 256, 0, stream>>>(w, wpk);
    pad_convert<<<(N_IMG * H_PAD * H_PAD * C_IN) / 256, 256, 0, stream>>>(x, xp);

    bconv_gemm<<<PIX / 64, 256, 0, stream>>>(xp, wpk, b, out);
}
